// LorentzTransformerEncoder_21199958573248
// MI455X (gfx1250) — compile-verified, matrix operand order checked by reading, not run on hardware
//
#include <hip/hip_runtime.h>
#include <hip/hip_bf16.h>
#include <math.h>

typedef __attribute__((ext_vector_type(16))) _Float16 v16h;
typedef __attribute__((ext_vector_type(8)))  _Float16 v8h;
typedef __attribute__((ext_vector_type(8)))  float    v8f;

// ---------------- problem dims ----------------
constexpr int Bb = 8, Nn = 1024, Dd = 769, Ds = 768, Hh = 12, HD = 64, Mm = 3072;
constexpr int Rr = Bb * Nn;        // 8192 rows
constexpr int KP = 800;            // 769 padded to mult of 32 (K dim)
constexpr int RP = 832;            // 769 padded to mult of 64 (weight rows / N dim)
constexpr int BH = Bb * Hh;        // 96
constexpr int QKP = 96;            // 65 (head dim + time) padded to mult of 32
constexpr int VTR = 80;            // 65 + ones-row, padded to mult of 16

// ---------------- workspace layout (bytes) ----------------
constexpr size_t SZ_W    = (size_t)RP * 800 * 2;           // one padded 769x769 weight (f16)
constexpr size_t OFF_WQ  = 0;
constexpr size_t OFF_WK  = OFF_WQ + SZ_W;
constexpr size_t OFF_WV  = OFF_WK + SZ_W;
constexpr size_t OFF_WO  = OFF_WV + SZ_W;
constexpr size_t OFF_W1  = OFF_WO + SZ_W;
constexpr size_t SZ_W1   = (size_t)3072 * 800 * 2;
constexpr size_t OFF_W2  = OFF_W1 + SZ_W1;
constexpr size_t SZ_W2   = (size_t)RP * 3072 * 2;
constexpr size_t OFF_H16 = OFF_W2 + SZ_W2;                 // h1 -> attn -> z  (f16, 8192x800)
constexpr size_t SZ_H16  = (size_t)Rr * KP * 2;
// region A (reused): Qf/Kf/Vf  ->  attn_c + o  ->  M2
constexpr size_t OFF_A   = OFF_H16 + SZ_H16;
constexpr size_t SZ_QF   = (size_t)Rr * Dd * 4;            // 8192x769 f32
constexpr size_t OFF_QF  = OFF_A;
constexpr size_t OFF_KF  = OFF_A + SZ_QF;
constexpr size_t OFF_VF  = OFF_A + 2 * SZ_QF;
constexpr size_t OFF_AC  = OFF_A;                          // attn_c: 96*1024*65 f32
constexpr size_t SZ_AC   = (size_t)BH * Nn * 65 * 4;
constexpr size_t OFF_O   = OFF_A + SZ_AC;                  // o: 8192x769 f32
constexpr size_t OFF_M2  = OFF_A;                          // M2: 8192x769 f32
// region B (reused): Qh/Kh/Vt  ->  G16 (FFN hidden)
constexpr size_t OFF_BB  = OFF_A + 3 * SZ_QF;
constexpr size_t SZ_QH   = (size_t)BH * Nn * QKP * 2;      // 96*1024*96 f16
constexpr size_t OFF_QH  = OFF_BB;
constexpr size_t OFF_KH  = OFF_BB + SZ_QH;
constexpr size_t OFF_VT  = OFF_BB + 2 * SZ_QH;             // 96*80*1024 f16
constexpr size_t SZ_VT   = (size_t)BH * VTR * Nn * 2;
constexpr size_t OFF_G16 = OFF_BB;                         // 8192x3072 f16
constexpr size_t OFF_OUT1 = OFF_BB + 2 * SZ_QH + SZ_VT;    // 8192x768 f32, persistent
// total ~183 MB

// ---------------- helpers ----------------
__device__ inline v16h pack16(v8h lo, v8h hi) {
  v16h r;
#pragma unroll
  for (int i = 0; i < 8; ++i) { r[i] = lo[i]; r[8 + i] = hi[i]; }
  return r;
}
// Per-lane fragment load for A (16x32 MxK) and B (32x16 KxN viewed as Bt rows):
// lane holds row (lane&15), K halves {kb..kb+7, kb+16..kb+23}, kb = (lane>>4)*8.
// Caller passes p already offset by row*ld + kb.
__device__ inline v16h load_frag(const _Float16* p) {
  v8h lo = *(const v8h*)p;
  v8h hi = *(const v8h*)(p + 16);
  return pack16(lo, hi);
}

__device__ inline float block_sum(float v, float* red) {
  int tid = threadIdx.x;
  red[tid] = v;
  __syncthreads();
  for (int off = 128; off > 0; off >>= 1) {
    if (tid < off) red[tid] += red[tid + off];
    __syncthreads();
  }
  float r = red[0];
  __syncthreads();
  return r;
}

__device__ inline float gelu_tanh(float x) {
  return 0.5f * x * (1.f + tanhf(0.7978845608028654f * (x + 0.044715f * x * x * x)));
}

// ---------------- weight conversion: f32 -> padded f16 ----------------
__global__ void cvt_w_kernel(const float* __restrict__ src, _Float16* __restrict__ dst,
                             int rows, int cols, int rowsP, int colsP) {
  int gid = blockIdx.x * blockDim.x + threadIdx.x;
  int total = rowsP * colsP;
  if (gid >= total) return;
  int r = gid / colsP, c = gid - r * colsP;
  float v = (r < rows && c < cols) ? src[(size_t)r * cols + c] : 0.f;
  dst[gid] = (_Float16)v;
}

// ---------------- LN1: x (f32, 769 w/ time col) -> h1 (f16, 800) ----------------
__global__ __launch_bounds__(256) void ln1_kernel(const float* __restrict__ x,
                                                  const float* __restrict__ gamma,
                                                  const float* __restrict__ beta,
                                                  _Float16* __restrict__ h16) {
  int row = blockIdx.x, tid = threadIdx.x;
  __shared__ float red[256];
  const float* xr = x + (size_t)row * Dd + 1;
  float s = 0.f;
  for (int j = tid; j < Ds; j += 256) s += xr[j];
  float mu = block_sum(s, red) * (1.f / Ds);
  float v = 0.f;
  for (int j = tid; j < Ds; j += 256) { float d = xr[j] - mu; v += d * d; }
  float var = block_sum(v, red) * (1.f / Ds);
  float inv = rsqrtf(var + 1e-5f);
  _Float16* hr = h16 + (size_t)row * KP;
  float sq = 0.f;
  for (int j = tid; j < Ds; j += 256) {
    float xn = gamma[j] * (xr[j] - mu) * inv + beta[j];
    hr[1 + j] = (_Float16)xn;
    sq += xn * xn;
  }
  float tot = block_sum(sq, red);
  if (tid == 0) hr[0] = (_Float16)sqrtf(1.f + tot);
  for (int j = Dd + tid; j < KP; j += 256) hr[j] = (_Float16)0.f;
}

// ---------------- register-blocked WMMA GEMM: C[M,ncols] = A[M,K] @ W[ncols,K]^T --------
// Each wave computes a 64x64 tile (4x4 WMMA tiles, 16 f32 accumulators = 128 VGPRs).
// 16 WMMAs per 16 b128 fragment loads -> ~32 flops/byte of L2 traffic.
// mode 0: store f32 to Cf ; mode 1: store gelu(val) as f16 to Ch
__global__ __launch_bounds__(128) void gemm_f16_wmma(const _Float16* __restrict__ A, int lda,
                                                     const _Float16* __restrict__ Bw, int ldb,
                                                     float* __restrict__ Cf,
                                                     _Float16* __restrict__ Ch, int ldc,
                                                     int ncols, int ktiles, int mode) {
  int wave = threadIdx.x >> 5;
  int lane = threadIdx.x & 31;
  int nblk = blockIdx.x * 4 + wave;                 // 64-col block
  int nblks = (ncols + 63) >> 6;
  if (nblk >= nblks) return;
  int mblk = blockIdx.y;                            // 64-row block
  int row16 = lane & 15, hl = lane >> 4;
  int khalf = hl * 8;

  const _Float16* Ap = A + (size_t)(mblk * 64 + row16) * lda + khalf;
  const _Float16* Bp = Bw + (size_t)(nblk * 64 + row16) * ldb + khalf;
  size_t astep = (size_t)16 * lda;                  // next 16-row tile
  size_t bstep = (size_t)16 * ldb;

  v8f acc[4][4] = {};
  for (int kt = 0; kt < ktiles; ++kt) {
    int k0 = kt * 32;
    __builtin_prefetch(Ap + k0 + 256, 0, 1);
    __builtin_prefetch(Bp + k0 + 256, 0, 1);
    v16h af[4], bf[4];
#pragma unroll
    for (int i = 0; i < 4; ++i) af[i] = load_frag(Ap + (size_t)i * astep + k0);
#pragma unroll
    for (int j = 0; j < 4; ++j) bf[j] = load_frag(Bp + (size_t)j * bstep + k0);
#pragma unroll
    for (int i = 0; i < 4; ++i)
#pragma unroll
      for (int j = 0; j < 4; ++j)
        acc[i][j] = __builtin_amdgcn_wmma_f32_16x16x32_f16(false, af[i], false, bf[j],
                                                           (short)0, acc[i][j], false, false);
  }

#pragma unroll
  for (int j = 0; j < 4; ++j) {
    int col = nblk * 64 + j * 16 + row16;
    if (col >= ncols) continue;
#pragma unroll
    for (int i = 0; i < 4; ++i) {
      size_t rowb = (size_t)(mblk * 64 + i * 16 + hl * 8);
      if (mode == 0) {
#pragma unroll
        for (int r = 0; r < 8; ++r)
          Cf[(rowb + r) * ldc + col] = acc[i][j][r];
      } else {
#pragma unroll
        for (int r = 0; r < 8; ++r)
          Ch[(rowb + r) * ldc + col] = (_Float16)gelu_tanh(acc[i][j][r]);
      }
    }
  }
}

// ---------------- split heads: Qf/Kf/Vf (f32 8192x769) -> Qh/Kh (f16 [bh,n,96]), Vt (f16 [bh,80,n]) ----
__global__ void split_heads_kernel(const float* __restrict__ Qf, const float* __restrict__ Kf,
                                   const float* __restrict__ Vf,
                                   _Float16* __restrict__ Qh, _Float16* __restrict__ Kh,
                                   _Float16* __restrict__ Vt) {
  int gid = blockIdx.x * blockDim.x + threadIdx.x; // (b*H+h)*N + n
  if (gid >= BH * Nn) return;
  int n = gid & (Nn - 1);
  int bh = gid >> 10;
  int h = bh % Hh;
  int b = bh / Hh;
  size_t rowoff = ((size_t)b * Nn + n) * Dd + 1 + h * HD;

  _Float16* qo = Qh + (size_t)gid * QKP;
  float sq = 0.f;
  for (int d = 0; d < HD; ++d) { float v = Qf[rowoff + d]; sq += v * v; qo[1 + d] = (_Float16)v; }
  qo[0] = (_Float16)sqrtf(1.f + sq);
  for (int d = 65; d < QKP; ++d) qo[d] = (_Float16)0.f;

  _Float16* ko = Kh + (size_t)gid * QKP;
  float sk = 0.f;
  for (int d = 0; d < HD; ++d) { float v = Kf[rowoff + d]; sk += v * v; ko[1 + d] = (_Float16)(-v); }
  ko[0] = (_Float16)sqrtf(1.f + sk);
  for (int d = 65; d < QKP; ++d) ko[d] = (_Float16)0.f;

  _Float16* vo = Vt + (size_t)bh * VTR * Nn;
  float sv = 0.f;
  for (int d = 0; d < HD; ++d) { float v = Vf[rowoff + d]; sv += v * v; vo[(size_t)(1 + d) * Nn + n] = (_Float16)v; }
  vo[n] = (_Float16)sqrtf(1.f + sv);       // row 0: v time
  vo[(size_t)65 * Nn + n] = (_Float16)1.f; // ones row -> softmax denominator
  for (int d = 66; d < VTR; ++d) vo[(size_t)d * Nn + n] = (_Float16)0.f;
}

// ---------------- fused Lorentz attention, one wave per (bh, 16-row i-tile) ----------------
// out row i: sum_j exp(f(K'_i . Q'_j)) * V[j] / den, then hyperboloid renorm -> attn_c [bh,n,65] f32
__global__ __launch_bounds__(32) void attn_kernel(const _Float16* __restrict__ Qh,
                                                  const _Float16* __restrict__ Kh,
                                                  const _Float16* __restrict__ Vt,
                                                  float* __restrict__ attn_c) {
  int bh = blockIdx.y;
  int it = blockIdx.x;
  int lane = threadIdx.x;
  int row16 = lane & 15, hl = lane >> 4;
  int khalf = hl * 8;

  __shared__ __align__(16) _Float16 Pls[16][40];
  __shared__ float S2[16][84];

  // A fragments from K' (output index side), preloaded for all 3 K-tiles
  const _Float16* Kp = Kh + ((size_t)bh * Nn + it * 16 + row16) * QKP + khalf;
  v16h ak0 = load_frag(Kp);
  v16h ak1 = load_frag(Kp + 32);
  v16h ak2 = load_frag(Kp + 64);

  v8f acc0 = {}, acc1 = {}, acc2 = {}, acc3 = {}, acc4 = {};
  const _Float16* Qbase = Qh + (size_t)bh * Nn * QKP;
  const _Float16* Vbase = Vt + (size_t)bh * VTR * Nn;

  for (int jt = 0; jt < Nn / 32; ++jt) {
    int j0 = jt * 32;
#pragma unroll
    for (int jj = 0; jj < 2; ++jj) {
      const _Float16* Qp = Qbase + (size_t)(j0 + jj * 16 + row16) * QKP + khalf;
      v8f s = {};
      s = __builtin_amdgcn_wmma_f32_16x16x32_f16(false, ak0, false, load_frag(Qp), (short)0, s, false, false);
      s = __builtin_amdgcn_wmma_f32_16x16x32_f16(false, ak1, false, load_frag(Qp + 32), (short)0, s, false, false);
      s = __builtin_amdgcn_wmma_f32_16x16x32_f16(false, ak2, false, load_frag(Qp + 64), (short)0, s, false, false);
#pragma unroll
      for (int r = 0; r < 8; ++r) {
        float d2 = fmaxf(2.f * (s[r] - 1.f), 1e-8f);
        float f = 1.f / (1.f + log1pf(d2));
        Pls[r + hl * 8][jj * 16 + row16] = (_Float16)expf(f); // f in (0,1]: no max-sub needed
      }
    }
    __syncthreads();
    v16h ap = load_frag(&Pls[row16][khalf]); // P as A-fragment (K = j within block)
#pragma unroll
    for (int nt = 0; nt < 5; ++nt) {
      const _Float16* Vp = Vbase + (size_t)(nt * 16 + row16) * Nn + j0 + khalf;
      v16h bv = load_frag(Vp);
      v8f* ac = nt == 0 ? &acc0 : nt == 1 ? &acc1 : nt == 2 ? &acc2 : nt == 3 ? &acc3 : &acc4;
      *ac = __builtin_amdgcn_wmma_f32_16x16x32_f16(false, ap, false, bv, (short)0, *ac, false, false);
    }
    __syncthreads();
  }

#pragma unroll
  for (int r = 0; r < 8; ++r) {
    S2[r + hl * 8][0 * 16 + row16] = acc0[r];
    S2[r + hl * 8][1 * 16 + row16] = acc1[r];
    S2[r + hl * 8][2 * 16 + row16] = acc2[r];
    S2[r + hl * 8][3 * 16 + row16] = acc3[r];
    S2[r + hl * 8][4 * 16 + row16] = acc4[r];
  }
  __syncthreads();
  if (lane < 16) {
    float inv = 1.f / S2[lane][65];          // denominator (ones row)
    float s0 = S2[lane][0] * inv;
    float l = s0 * s0;
    for (int d = 1; d < 65; ++d) { float sd = S2[lane][d] * inv; l -= sd * sd; }
    float sc = rsqrtf(fmaxf(l, 1e-8f));      // sqrt(K)=1
    float* outp = attn_c + ((size_t)bh * Nn + it * 16 + lane) * 65;
    outp[0] = s0 * sc;
    for (int d = 1; d < 65; ++d) outp[d] = S2[lane][d] * inv * sc;
  }
}

// ---------------- merge heads: attn_c -> attn16 (f16 8192x800) ----------------
__global__ __launch_bounds__(256) void merge_heads_kernel(const float* __restrict__ attn_c,
                                                          _Float16* __restrict__ attn16) {
  int row = blockIdx.x * 8 + (threadIdx.x >> 5);
  int lane = threadIdx.x & 31;
  int b = row >> 10, n = row & (Nn - 1);
  float ts = 0.f;
  for (int h = 0; h < Hh; ++h) {
    float ct = attn_c[(((size_t)(b * Hh + h)) * Nn + n) * 65];
    ts += ct * ct;
  }
  _Float16* ar = attn16 + (size_t)row * KP;
  if (lane == 0) ar[0] = (_Float16)sqrtf(ts - (float)(Hh - 1));
  for (int idx = lane; idx < Ds; idx += 32) {
    int h = idx >> 6, d = idx & 63;
    ar[1 + idx] = (_Float16)attn_c[(((size_t)(b * Hh + h)) * Nn + n) * 65 + 1 + d];
  }
  for (int idx = Dd + lane; idx < KP; idx += 32) ar[idx] = (_Float16)0.f;
}

// ---------------- residual + LN2: o,x -> out1 (f32 768), z (f16 800) ----------------
__global__ __launch_bounds__(256) void resid_ln2_kernel(const float* __restrict__ o,
                                                        const float* __restrict__ x,
                                                        const float* __restrict__ gamma,
                                                        const float* __restrict__ beta,
                                                        float* __restrict__ out1,
                                                        _Float16* __restrict__ z16) {
  int row = blockIdx.x, tid = threadIdx.x;
  __shared__ float red[256];
  const float* orow = o + (size_t)row * Dd + 1;
  const float* xrow = x + (size_t)row * Dd + 1;
  float* o1 = out1 + (size_t)row * Ds;
  float s = 0.f;
  for (int j = tid; j < Ds; j += 256) { float v = orow[j] + xrow[j]; o1[j] = v; s += v; }
  float mu = block_sum(s, red) * (1.f / Ds);
  float v2 = 0.f;
  for (int j = tid; j < Ds; j += 256) { float d = o1[j] - mu; v2 += d * d; }
  float var = block_sum(v2, red) * (1.f / Ds);
  float inv = rsqrtf(var + 1e-5f);
  _Float16* zr = z16 + (size_t)row * KP;
  float sq = 0.f;
  for (int j = tid; j < Ds; j += 256) {
    float xn = gamma[j] * (o1[j] - mu) * inv + beta[j];
    zr[1 + j] = (_Float16)xn;
    sq += xn * xn;
  }
  float tot = block_sum(sq, red);
  if (tid == 0) zr[0] = (_Float16)sqrtf(1.f + tot);
  for (int j = Dd + tid; j < KP; j += 256) zr[j] = (_Float16)0.f;
}

// ---------------- FFN hidden time column: G16[:,0] = sqrt(1 + sum_{j>=1} g^2) ----------------
__global__ __launch_bounds__(256) void ffn_time_kernel(_Float16* __restrict__ G16) {
  int row = blockIdx.x, tid = threadIdx.x;
  __shared__ float red[256];
  _Float16* g = G16 + (size_t)row * Mm;
  float sq = 0.f;
  for (int j = 1 + tid; j < Mm; j += 256) { float v = (float)g[j]; sq += v * v; }
  float tot = block_sum(sq, red);
  if (tid == 0) g[0] = (_Float16)sqrtf(1.f + tot);
}

// ---------------- final: out = add_time(M2[:,1:] + out1) ----------------
__global__ __launch_bounds__(256) void final_kernel(const float* __restrict__ M2,
                                                    const float* __restrict__ out1,
                                                    float* __restrict__ out) {
  int row = blockIdx.x, tid = threadIdx.x;
  __shared__ float red[256];
  const float* m = M2 + (size_t)row * Dd + 1;
  const float* o1 = out1 + (size_t)row * Ds;
  float* orow = out + (size_t)row * Dd;
  float sq = 0.f;
  for (int j = tid; j < Ds; j += 256) {
    float v = m[j] + o1[j];
    orow[1 + j] = v;
    sq += v * v;
  }
  float tot = block_sum(sq, red);
  if (tid == 0) orow[0] = sqrtf(1.f + tot);
}

// ---------------- launch ----------------
extern "C" void kernel_launch(void* const* d_in, const int* in_sizes, int n_in,
                              void* d_out, int out_size, void* d_ws, size_t ws_size,
                              hipStream_t stream) {
  const float* x      = (const float*)d_in[0];
  const float* gamma1 = (const float*)d_in[1];
  const float* beta1  = (const float*)d_in[2];
  const float* Wq     = (const float*)d_in[3];
  const float* Wk     = (const float*)d_in[4];
  const float* Wv     = (const float*)d_in[5];
  const float* Wo     = (const float*)d_in[6];
  const float* gamma2 = (const float*)d_in[7];
  const float* beta2  = (const float*)d_in[8];
  const float* W1     = (const float*)d_in[9];
  const float* W2     = (const float*)d_in[10];
  float* out = (float*)d_out;
  char* ws = (char*)d_ws;

  _Float16* WQ16 = (_Float16*)(ws + OFF_WQ);
  _Float16* WK16 = (_Float16*)(ws + OFF_WK);
  _Float16* WV16 = (_Float16*)(ws + OFF_WV);
  _Float16* WO16 = (_Float16*)(ws + OFF_WO);
  _Float16* W116 = (_Float16*)(ws + OFF_W1);
  _Float16* W216 = (_Float16*)(ws + OFF_W2);
  _Float16* H16  = (_Float16*)(ws + OFF_H16);   // h1 -> attn -> z
  float*    Qf   = (float*)(ws + OFF_QF);
  float*    Kf   = (float*)(ws + OFF_KF);
  float*    Vf   = (float*)(ws + OFF_VF);
  float*    Ac   = (float*)(ws + OFF_AC);
  float*    Of   = (float*)(ws + OFF_O);
  float*    M2   = (float*)(ws + OFF_M2);
  _Float16* Qh   = (_Float16*)(ws + OFF_QH);
  _Float16* Khb  = (_Float16*)(ws + OFF_KH);
  _Float16* Vt   = (_Float16*)(ws + OFF_VT);
  _Float16* G16  = (_Float16*)(ws + OFF_G16);
  float*    Out1 = (float*)(ws + OFF_OUT1);

  auto cdiv = [](int a, int b) { return (a + b - 1) / b; };

  // 1. convert weights to padded f16 (rows padded to mult of 64 for 64-wide N blocks)
  cvt_w_kernel<<<cdiv(RP * 800, 256), 256, 0, stream>>>(Wq, WQ16, Dd, Dd, RP, 800);
  cvt_w_kernel<<<cdiv(RP * 800, 256), 256, 0, stream>>>(Wk, WK16, Dd, Dd, RP, 800);
  cvt_w_kernel<<<cdiv(RP * 800, 256), 256, 0, stream>>>(Wv, WV16, Dd, Dd, RP, 800);
  cvt_w_kernel<<<cdiv(RP * 800, 256), 256, 0, stream>>>(Wo, WO16, Dd, Dd, RP, 800);
  cvt_w_kernel<<<cdiv(3072 * 800, 256), 256, 0, stream>>>(W1, W116, Mm, Dd, Mm, 800);
  cvt_w_kernel<<<cdiv(RP * 3072, 256), 256, 0, stream>>>(W2, W216, Dd, Mm, RP, Mm);

  // 2. LN1
  ln1_kernel<<<Rr, 256, 0, stream>>>(x, gamma1, beta1, H16);

  // 3. Q/K/V projections (f32 out); each wave does a 64x64 tile, block = 4 waves = 256 cols
  dim3 gproj(cdiv(cdiv(Dd, 64), 4), Rr / 64);
  gemm_f16_wmma<<<gproj, 128, 0, stream>>>(H16, KP, WQ16, KP, Qf, nullptr, Dd, Dd, KP / 32, 0);
  gemm_f16_wmma<<<gproj, 128, 0, stream>>>(H16, KP, WK16, KP, Kf, nullptr, Dd, Dd, KP / 32, 0);
  gemm_f16_wmma<<<gproj, 128, 0, stream>>>(H16, KP, WV16, KP, Vf, nullptr, Dd, Dd, KP / 32, 0);

  // 4. split heads (+ per-head time, k-negation, V transpose w/ ones-row)
  split_heads_kernel<<<cdiv(BH * Nn, 256), 256, 0, stream>>>(Qf, Kf, Vf, Qh, Khb, Vt);

  // 5. fused attention  (attn_c overwrites Qf region - Qf/Kf/Vf are dead)
  attn_kernel<<<dim3(Nn / 16, BH), 32, 0, stream>>>(Qh, Khb, Vt, Ac);

  // 6. merge heads (writes into H16; h1 is dead)
  merge_heads_kernel<<<Rr / 8, 256, 0, stream>>>(Ac, H16);

  // 7. O projection
  gemm_f16_wmma<<<gproj, 128, 0, stream>>>(H16, KP, WO16, KP, Of, nullptr, Dd, Dd, KP / 32, 0);

  // 8. residual + LN2 (writes out1 + z into H16)
  resid_ln2_kernel<<<Rr, 256, 0, stream>>>(Of, x, gamma2, beta2, Out1, H16);

  // 9. FFN GEMM1 with fused GELU -> f16 hidden (G16 overwrites Qh/Kh/Vt region)
  dim3 g1(cdiv(Mm / 64, 4), Rr / 64);
  gemm_f16_wmma<<<g1, 128, 0, stream>>>(H16, KP, W116, KP, nullptr, G16, Mm, Mm, KP / 32, 1);

  // 10. hidden time column
  ffn_time_kernel<<<Rr, 256, 0, stream>>>(G16);

  // 11. FFN GEMM2 (M2 overwrites attn_c region)
  gemm_f16_wmma<<<gproj, 128, 0, stream>>>(G16, Mm, W216, Mm, M2, nullptr, Dd, Dd, Mm / 32, 0);

  // 12. final residual + add_time
  final_kernel<<<Rr, 256, 0, stream>>>(M2, Out1, out);
}
